// MotionPredictor_11347303596287
// MI455X (gfx1250) — compile-verified, matrix operand order checked by reading, not run on hardware
//
#include <hip/hip_runtime.h>
#include <cstdint>

// ---------------------------------------------------------------------------
// GRU encoder-decoder, MI455X (gfx1250, wave32, WMMA 16x16x32 bf16).
// B=4096, S-1=49 enc steps, T=25 dec steps, H=1024, I=69 (pad 96), P=10.
// Matrix-core bound (~8.4 TFLOP GEMM, weights L2-resident). Weights are
// staged to LDS with gfx1250 async global->LDS copies (ASYNCcnt), the fp32
// h activations are converted to bf16 during vectorized LDS staging.
// ---------------------------------------------------------------------------

typedef __attribute__((ext_vector_type(16))) __bf16 v16bf;
typedef __attribute__((ext_vector_type(8)))  float  v8f;

#define KLDS 104   // 96 + 8 pad (208B row stride, 16B aligned, bank-skewed)
#define KLDS2 72   // 64 + 8 pad for the fc kernel

static constexpr int Hh  = 1024;
static constexpr int KX  = 96;     // padded input dim
static constexpr int NI  = 69;
static constexpr int NFC = 80;     // padded fc rows
static constexpr int TT  = 25;
static constexpr int BB  = 4096;

// --- gfx1250 async global->LDS copy (16B per lane), tracked by ASYNCcnt ---
__device__ __forceinline__ void async_cp16(const void* gptr, void* lptr) {
  unsigned lds = (unsigned)(uintptr_t)lptr;   // addr[31:0] == LDS offset
  asm volatile("global_load_async_to_lds_b128 %0, %1, off"
               :: "v"(lds), "v"(gptr) : "memory");
}
__device__ __forceinline__ void async_wait0() {
  asm volatile("s_wait_asynccnt 0x0" ::: "memory");
}

// Pack 4 fp32 -> 4 bf16 and store 8B to LDS.
__device__ __forceinline__ void pack4_bf16(__bf16* dst, float4 f) {
  union { __bf16 b[4]; uint2 u; } pk;
  pk.b[0] = (__bf16)f.x; pk.b[1] = (__bf16)f.y;
  pk.b[2] = (__bf16)f.z; pk.b[3] = (__bf16)f.w;
  *(uint2*)dst = pk.u;
}

// Load a 16x32 bf16 WMMA fragment from an LDS row (row-major, any stride).
// ISA layout: lanes 0-15 hold K 0..7 & 16..23; lanes 16-31 hold K 8..15 & 24..31.
__device__ __forceinline__ v16bf frag_ld(const __bf16* row, int kk, int lh) {
  const __bf16* p = row + kk + lh;
  v16bf f;
#pragma unroll
  for (int i = 0; i < 8; ++i) f[i] = p[i];
#pragma unroll
  for (int i = 0; i < 8; ++i) f[8 + i] = p[16 + i];
  return f;
}

__device__ __forceinline__ v8f wmma_bf16(v16bf a, v16bf b, v8f c) {
  return __builtin_amdgcn_wmma_f32_16x16x32_bf16(false, a, false, b, (short)0, c,
                                                 false, false);
}

// One A-tile x three gate B-tiles, KC columns of K, accumulate r/z/n sets.
__device__ __forceinline__ void mma_phase(
    const __bf16 (*As)[KLDS],
    const __bf16 (*B0)[KLDS], const __bf16 (*B1)[KLDS], const __bf16 (*B2)[KLDS],
    int KC, int lane, int wrow, int wcol, v8f* ar, v8f* az, v8f* an) {
  const int rr = wrow * 16 + (lane & 15);
  const int lh = (lane >> 4) * 8;
  for (int kk = 0; kk < KC; kk += 32) {
    v16bf a = frag_ld(As[rr], kk, lh);
#pragma unroll
    for (int t = 0; t < 2; ++t) {
      const int bn = wcol * 32 + t * 16 + (lane & 15);
      v16bf br = frag_ld(B0[bn], kk, lh);
      v16bf bz = frag_ld(B1[bn], kk, lh);
      v16bf bq = frag_ld(B2[bn], kk, lh);
      ar[t] = wmma_bf16(a, br, ar[t]);
      az[t] = wmma_bf16(a, bz, az[t]);
      an[t] = wmma_bf16(a, bq, an[t]);
    }
  }
}

// Fused GRU step: gates GEMM (x@Wih^T + h@Whh^T) + activation -> h_next.
// Block tile: 64 rows x 64 hidden cols; 8 waves in a 4x2 grid, each 16x32.
__global__ __launch_bounds__(256) void gru_gate_kernel(
    const float* __restrict__ x, long long x_stride, int x_cols,
    const float* __restrict__ h_prev, float* __restrict__ h_next,
    const __bf16* __restrict__ Wih, const __bf16* __restrict__ Whh,
    const float* __restrict__ b_ih, const float* __restrict__ b_hh) {
  __shared__ __bf16 As[64][KLDS];
  __shared__ __bf16 Bs[3][64][KLDS];

  const int m0 = blockIdx.x * 64;
  const int n0 = blockIdx.y * 64;
  const int tid = threadIdx.x;
  const int lane = tid & 31;
  const int wid = tid >> 5;
  const int wrow = wid >> 1, wcol = wid & 1;

  v8f accr[2] = {}; v8f accz[2] = {}; v8f accnx[2] = {}; v8f accnh[2] = {};

  // ---- x phase: K = 96 (padded, cols >= x_cols are zero) ----
  // A: scalar staging (encoder row stride 3381 floats is not 16B aligned).
  for (int idx = tid; idx < 64 * 96; idx += 256) {
    const int r = idx / 96, k = idx % 96;
    float v = (k < x_cols) ? x[(long long)(m0 + r) * x_stride + k] : 0.f;
    As[r][k] = (__bf16)v;
  }
  // B: async copy, 3 gates x 64 rows x 12 16B-chunks = 2304 chunks (9 iters).
  for (int c = tid; c < 3 * 64 * 12; c += 256) {
    const int g = c / (64 * 12);
    const int r = (c / 12) & 63;
    const int kc = (c % 12) * 8;
    async_cp16(Wih + (size_t)(g * Hh + n0 + r) * KX + kc, &Bs[g][r][kc]);
  }
  async_wait0();
  __syncthreads();
  mma_phase(As, Bs[0], Bs[1], Bs[2], 96, lane, wrow, wcol, accr, accz, accnx);
  __syncthreads();

  // ---- h phase: K = 1024 in 16 chunks of 64 ----
  for (int c = 0; c < 16; ++c) {
    const int k0 = c * 64;
    // A: float4 loads + packed bf16 stores (rows 16B aligned: stride 1024 f32).
    for (int idx = tid; idx < 64 * 16; idx += 256) {
      const int r = idx >> 4, c4 = (idx & 15) << 2;
      float4 f = *(const float4*)&h_prev[(size_t)(m0 + r) * Hh + k0 + c4];
      pack4_bf16(&As[r][c4], f);
    }
    // B: async global->LDS, 3 x 64 x 8 chunks = 1536 (6 iters).
    for (int cc = tid; cc < 3 * 64 * 8; cc += 256) {
      const int g = cc >> 9;
      const int r = (cc >> 3) & 63;
      const int kc = (cc & 7) * 8;
      async_cp16(Whh + (size_t)(g * Hh + n0 + r) * Hh + k0 + kc, &Bs[g][r][kc]);
    }
    if (c + 1 < 16)  // gfx1250 global_prefetch_b8 for next weight chunk
      __builtin_prefetch(Whh + (size_t)n0 * Hh + (size_t)(c + 1) * 64, 0, 0);
    async_wait0();
    __syncthreads();
    mma_phase(As, Bs[0], Bs[1], Bs[2], 64, lane, wrow, wcol, accr, accz, accnh);
    __syncthreads();
  }

  // ---- epilogue: GRU nonlinearity in registers, write h_next (fp32) ----
#pragma unroll
  for (int t = 0; t < 2; ++t) {
    const int col = n0 + wcol * 32 + t * 16 + (lane & 15);
    const float br_ = b_ih[col] + b_hh[col];
    const float bz_ = b_ih[Hh + col] + b_hh[Hh + col];
    const float bnx_ = b_ih[2 * Hh + col];
    const float bnh_ = b_hh[2 * Hh + col];
    const int rbase = m0 + wrow * 16 + ((lane >> 4) << 3);
#pragma unroll
    for (int v = 0; v < 8; ++v) {
      const int row = rbase + v;
      const float r = 1.f / (1.f + __expf(-(accr[t][v] + br_)));
      const float z = 1.f / (1.f + __expf(-(accz[t][v] + bz_)));
      const float nn = tanhf(accnx[t][v] + bnx_ + r * (accnh[t][v] + bnh_));
      const float hp = h_prev[(size_t)row * Hh + col];
      h_next[(size_t)row * Hh + col] = (1.f - z) * nn + z * hp;
    }
  }
}

// Decoder output: out = inp + h @ Wfc^T + b_fc; writes d_out and next inp.
// Block: 128 rows x 80 cols (5 N-tiles), 8 waves of 16 rows each.
__global__ __launch_bounds__(256) void dec_out_kernel(
    const float* __restrict__ h, const float* __restrict__ inp,
    float* __restrict__ inp_next, float* __restrict__ out,
    const __bf16* __restrict__ Wfc, const float* __restrict__ b_fc, int tstep) {
  __shared__ __bf16 As[128][KLDS2];
  __shared__ __bf16 Bs[NFC][KLDS2];
  const int m0 = blockIdx.x * 128;
  const int tid = threadIdx.x, lane = tid & 31, wid = tid >> 5;
  v8f acc[5] = {};
  for (int c = 0; c < 16; ++c) {
    const int k0 = c * 64;
    // A: vectorized fp32->bf16 staging, 128 x 16 float4s (8 iters).
    for (int idx = tid; idx < 128 * 16; idx += 256) {
      const int r = idx >> 4, c4 = (idx & 15) << 2;
      float4 f = *(const float4*)&h[(size_t)(m0 + r) * Hh + k0 + c4];
      pack4_bf16(&As[r][c4], f);
    }
    // B: async global->LDS, 80 x 8 chunks = 640.
    for (int cc = tid; cc < NFC * 8; cc += 256) {
      const int r = cc >> 3;
      const int kc = (cc & 7) * 8;
      async_cp16(Wfc + (size_t)r * Hh + k0 + kc, &Bs[r][kc]);
    }
    async_wait0();
    __syncthreads();
    const int rr = wid * 16 + (lane & 15);
    const int lh = (lane >> 4) * 8;
    for (int kk = 0; kk < 64; kk += 32) {
      v16bf a = frag_ld(As[rr], kk, lh);
#pragma unroll
      for (int t = 0; t < 5; ++t) {
        v16bf b = frag_ld(Bs[t * 16 + (lane & 15)], kk, lh);
        acc[t] = wmma_bf16(a, b, acc[t]);
      }
    }
    __syncthreads();
  }
#pragma unroll
  for (int t = 0; t < 5; ++t) {
    const int col = t * 16 + (lane & 15);
    const int rbase = m0 + wid * 16 + ((lane >> 4) << 3);
#pragma unroll
    for (int v = 0; v < 8; ++v) {
      const int row = rbase + v;
      if (col < NI) {
        const float val = acc[t][v] + b_fc[col] + inp[(size_t)row * NFC + col];
        out[((size_t)row * TT + tstep) * NI + col] = val;
        inp_next[(size_t)row * NFC + col] = val;
      }
    }
  }
}

// ---------------- setup / conversion kernels ----------------
__global__ void k_zero(float* p, long long n) {
  for (long long i = blockIdx.x * 256LL + threadIdx.x; i < n;
       i += (long long)gridDim.x * 256) p[i] = 0.f;
}
__global__ void k_cvt(const float* src, __bf16* dst, int rs, int cs, int rd, int cd) {
  const long long n = (long long)rd * cd;
  for (long long i = blockIdx.x * 256LL + threadIdx.x; i < n;
       i += (long long)gridDim.x * 256) {
    const int r = (int)(i / cd), k = (int)(i % cd);
    const float v = (r < rs && k < cs) ? src[(long long)r * cs + k] : 0.f;
    dst[i] = (__bf16)v;
  }
}
__global__ void k_dec_init_h(const float* hfin, const float* noise, float* hdec) {
  const long long n = (long long)10 * BB * Hh;
  for (long long i = blockIdx.x * 256LL + threadIdx.x; i < n;
       i += (long long)gridDim.x * 256) {
    const long long row = i >> 10;           // p*B + b
    const int j = (int)(i & 1023);
    const int b = (int)(row & (BB - 1));
    hdec[i] = hfin[(size_t)b * Hh + j] + noise[i];
  }
}
__global__ void k_dec_init_x(const float* dec_inputs, float* inpA, float* inpB) {
  const long long n = (long long)10 * BB * NFC;
  for (long long i = blockIdx.x * 256LL + threadIdx.x; i < n;
       i += (long long)gridDim.x * 256) {
    const long long row = i / NFC;
    const int k = (int)(i % NFC);
    const int b = (int)(row & (BB - 1));
    inpA[i] = (k < NI) ? dec_inputs[(size_t)b * TT * NI + k] : 0.f;
    inpB[i] = 0.f;
  }
}

extern "C" void kernel_launch(void* const* d_in, const int* in_sizes, int n_in,
                              void* d_out, int out_size, void* d_ws, size_t ws_size,
                              hipStream_t stream) {
  const float* enc   = (const float*)d_in[0];   // (B, 49, 69)
  const float* dec   = (const float*)d_in[1];   // (B, 25, 69)
  const float* W_ih  = (const float*)d_in[2];   // (3H, 69)
  const float* W_hh  = (const float*)d_in[3];   // (3H, H)
  const float* b_ih  = (const float*)d_in[4];
  const float* b_hh  = (const float*)d_in[5];
  const float* W_fc  = (const float*)d_in[6];   // (69, H)
  const float* b_fc  = (const float*)d_in[7];
  const float* noise = (const float*)d_in[8];   // (10, B, H)
  float* out = (float*)d_out;

  char* w = (char*)d_ws;
  auto alloc = [&](size_t bytes) -> void* {
    void* p = (void*)w;
    w += (bytes + 255) & ~(size_t)255;
    return p;
  };
  const long long MD = 10LL * BB;                        // 40960 decoder rows
  __bf16* Wihb = (__bf16*)alloc((size_t)3 * Hh * KX * 2);
  __bf16* Whhb = (__bf16*)alloc((size_t)3 * Hh * Hh * 2);
  __bf16* Wfcb = (__bf16*)alloc((size_t)NFC * Hh * 2);
  float* hA   = (float*)alloc((size_t)MD * Hh * 4);
  float* hB   = (float*)alloc((size_t)MD * Hh * 4);
  float* inpA = (float*)alloc((size_t)MD * NFC * 4);
  float* inpB = (float*)alloc((size_t)MD * NFC * 4);

  // weight conversion (+ zero-padding K->96, fc rows->80)
  k_cvt<<<1024, 256, 0, stream>>>(W_ih, Wihb, 3 * Hh, NI, 3 * Hh, KX);
  k_cvt<<<2048, 256, 0, stream>>>(W_hh, Whhb, 3 * Hh, Hh, 3 * Hh, Hh);
  k_cvt<<<512, 256, 0, stream>>>(W_fc, Wfcb, NI, Hh, NFC, Hh);

  // ---- encoder: h0 = 0, 49 fused GRU steps over batch 4096 ----
  k_zero<<<2048, 256, 0, stream>>>(hA, (long long)BB * Hh);
  float* hp = hA;
  float* hn = hB;
  for (int t = 0; t < 49; ++t) {
    gru_gate_kernel<<<dim3(BB / 64, Hh / 64), 256, 0, stream>>>(
        enc + (size_t)t * NI, 49LL * NI, NI, hp, hn, Wihb, Whhb, b_ih, b_hh);
    float* tmp = hp; hp = hn; hn = tmp;
  }
  // final encoder state is in hp

  // ---- decoder init: h = state + noise (all 10 paths), inp0 broadcast ----
  k_dec_init_h<<<8192, 256, 0, stream>>>(hp, noise, hn);
  k_dec_init_x<<<2048, 256, 0, stream>>>(dec, inpA, inpB);
  float* hdp = hn;   // decoder h_prev (40960 rows)
  float* hdn = hp;   // ping-pong target
  float* ip = inpA;
  float* in_ = inpB;

  // ---- decoder: 25 steps at effective batch 40960 ----
  for (int t = 0; t < TT; ++t) {
    gru_gate_kernel<<<dim3((unsigned)(MD / 64), Hh / 64), 256, 0, stream>>>(
        ip, NFC, NI, hdp, hdn, Wihb, Whhb, b_ih, b_hh);
    dec_out_kernel<<<(unsigned)(MD / 128), 256, 0, stream>>>(
        hdn, ip, in_, out, Wfcb, b_fc, t);
    float* tmp = hdp; hdp = hdn; hdn = tmp;
    tmp = ip; ip = in_; in_ = tmp;
  }
}
